// CAMSA_31370441130533
// MI455X (gfx1250) — matrix operand order repaired and checked by reading, not compiled, not run on hardware
//
#include <hip/hip_runtime.h>

// ---------------------------------------------------------------------------
// CAMSA for MI455X (gfx1250, wave32, WMMA).  All GEMMs via
// v_wmma_f32_16x16x32_bf16.  One NT GEMM building block: C = A * B^T where
// both A and B are stored row-major with K contiguous, matching the CDNA5
// 16-bit A-fragment (lane m=lane%16, K runs half*8..+7 and 16+half*8..+7) and
// B-fragment (lane n=lane%16, K runs half*8..+7 and 16+half*8..+7) VGPR layouts.
// Streaming (attn, 537MB) paths use non-temporal hints; bf16 tile staging
// uses GLOBAL_LOAD_ASYNC_TO_LDS_B128 when the builtin is available.
// ---------------------------------------------------------------------------

typedef __bf16 bf16_t;
typedef __attribute__((ext_vector_type(8)))  __bf16 v8bf;
typedef __attribute__((ext_vector_type(16))) __bf16 v16bf;
typedef __attribute__((ext_vector_type(8)))  float  v8f;
typedef __attribute__((ext_vector_type(4)))  float  fvec4;
typedef __attribute__((vector_size(16)))     int    i4x4;

struct FragBF { union { v16bf v; v8bf h[2]; }; };

#if __has_builtin(__builtin_amdgcn_global_load_async_to_lds_b128)
#define CAMSA_HAS_ASYNC 1
#else
#define CAMSA_HAS_ASYNC 0
#endif

__device__ __forceinline__ void camsa_async_b128(const void* g, void* l) {
#if CAMSA_HAS_ASYNC
  __builtin_amdgcn_global_load_async_to_lds_b128(
      (__attribute__((address_space(1))) i4x4*)(unsigned long long)(uintptr_t)g,
      (__attribute__((address_space(3))) i4x4*)(unsigned)(uintptr_t)l,
      0, 0);
#else
  (void)g; (void)l;
#endif
}

__device__ __forceinline__ void camsa_wait_async() {
#if CAMSA_HAS_ASYNC
#if __has_builtin(__builtin_amdgcn_s_wait_asynccnt)
  __builtin_amdgcn_s_wait_asynccnt(0);
#else
  asm volatile("s_wait_asynccnt 0" ::: "memory");
#endif
#endif
}

__device__ __forceinline__ v8bf camsa_cvt8(fvec4 a, fvec4 b) {
  v8bf h;
  h[0] = (__bf16)a.x; h[1] = (__bf16)a.y; h[2] = (__bf16)a.z; h[3] = (__bf16)a.w;
  h[4] = (__bf16)b.x; h[5] = (__bf16)b.y; h[6] = (__bf16)b.z; h[7] = (__bf16)b.w;
  return h;
}

// ---------------------------------------------------------------------------
// NT GEMM:  C[M,N] = alpha * A[M,K] * (B[N,K])^T (+ bias[N])
//   A: fp32 (converted to bf16 while staging) or bf16;  B: bf16;  C: f32/bf16
// Block: 256 threads = 8 waves, 4x2 wave grid.  BK = 32 (one WMMA K step).
// blockIdx.z batches heads: offset = (z/Hs)*s?b + (z%Hs)*s?h  (element units).
// ---------------------------------------------------------------------------
template <int BM, int BN, bool A_F32, bool OUT_BF16, bool NT_A, bool NT_C>
__global__ __launch_bounds__(256)
void camsa_gemm_nt(const void* __restrict__ Ap, const bf16_t* __restrict__ Bp,
                   void* __restrict__ Cp, const float* __restrict__ bias,
                   int lda, int ldb, int ldc, int K, int Hs,
                   long sAb, long sAh, long sBb, long sBh, long sCb, long sCh,
                   float alpha) {
  constexpr int BK   = 32;
  constexpr int LSTR = BK + 8;            // 40 bf16 = 80B row stride (16B-aligned, conflict-free)
  constexpr int WM   = BM / 4;            // wave tile rows
  constexpr int WN   = BN / 2;            // wave tile cols
  constexpr int MI   = WM / 16;
  constexpr int NI   = WN / 16;
  constexpr int AIT  = (BM * 4) / 256;    // 8-elem chunks per thread for A tile
  constexpr int BIT  = (BN * 4) / 256;

  __shared__ __align__(16) bf16_t Als[BM * LSTR];
  __shared__ __align__(16) bf16_t Bls[BN * LSTR];

  const int tid    = threadIdx.x;
  const int lane   = tid & 31;
  const int wave   = tid >> 5;
  const int wm     = wave >> 1;           // 0..3
  const int wn     = wave & 1;            // 0..1
  const int laneLo = lane & 15;
  const int laneHf = lane >> 4;

  const int z  = blockIdx.z;
  const int bz = z / Hs, hz = z % Hs;
  const char*   Ab = (const char*)Ap + (bz * sAb + hz * sAh) * (long)(A_F32 ? 4 : 2);
  const bf16_t* Bb = Bp + bz * sBb + hz * sBh;
  char*         Cb = (char*)Cp + (bz * sCb + hz * sCh) * (long)(OUT_BF16 ? 2 : 4);

  const int m0 = blockIdx.y * BM;
  const int n0 = blockIdx.x * BN;

  v8f acc[MI][NI] = {};

  const int kTiles = K / BK;
  for (int kt = 0; kt < kTiles; ++kt) {
    const int k0 = kt * BK;
    // ---- stage A tile (BM x 32) as bf16 ----
#pragma unroll
    for (int it = 0; it < AIT; ++it) {
      const int c = tid + it * 256;
      const int r = c >> 2, kc = (c & 3) * 8;
      if constexpr (A_F32) {
        const float* src = (const float*)Ab + (long)(m0 + r) * lda + k0 + kc;
        fvec4 f0, f1;
        if constexpr (NT_A) {
          f0 = __builtin_nontemporal_load((const fvec4*)src);
          f1 = __builtin_nontemporal_load((const fvec4*)(src + 4));
        } else {
          f0 = *(const fvec4*)src;
          f1 = *(const fvec4*)(src + 4);
          if (kt + 1 < kTiles) __builtin_prefetch(src + BK, 0, 1);  // global_prefetch_b8
        }
        *(v8bf*)&Als[r * LSTR + kc] = camsa_cvt8(f0, f1);
      } else {
        const bf16_t* src = (const bf16_t*)Ab + (long)(m0 + r) * lda + k0 + kc;
        if (CAMSA_HAS_ASYNC) {
          camsa_async_b128(src, &Als[r * LSTR + kc]);
        } else {
          *(v8bf*)&Als[r * LSTR + kc] = *(const v8bf*)src;
          if (kt + 1 < kTiles) __builtin_prefetch(src + BK, 0, 1);
        }
      }
    }
    // ---- stage B tile (BN x 32) ----
#pragma unroll
    for (int it = 0; it < BIT; ++it) {
      const int c = tid + it * 256;
      const int r = c >> 2, kc = (c & 3) * 8;
      const bf16_t* src = Bb + (long)(n0 + r) * ldb + k0 + kc;
      if (CAMSA_HAS_ASYNC) {
        camsa_async_b128(src, &Bls[r * LSTR + kc]);
      } else {
        *(v8bf*)&Bls[r * LSTR + kc] = *(const v8bf*)src;
        if (kt + 1 < kTiles) __builtin_prefetch(src + BK, 0, 1);
      }
    }
    camsa_wait_async();
    __syncthreads();

    // ---- fragments + WMMA ----
    FragBF a[MI], b[NI];
#pragma unroll
    for (int mi = 0; mi < MI; ++mi) {
      const int r = wm * WM + mi * 16 + laneLo;
      a[mi].h[0] = *(const v8bf*)&Als[r * LSTR + laneHf * 8];
      a[mi].h[1] = *(const v8bf*)&Als[r * LSTR + 16 + laneHf * 8];
    }
#pragma unroll
    for (int ni = 0; ni < NI; ++ni) {
      const int r = wn * WN + ni * 16 + laneLo;
      b[ni].h[0] = *(const v8bf*)&Bls[r * LSTR + laneHf * 8];
      b[ni].h[1] = *(const v8bf*)&Bls[r * LSTR + 16 + laneHf * 8];
    }
#pragma unroll
    for (int mi = 0; mi < MI; ++mi)
#pragma unroll
      for (int ni = 0; ni < NI; ++ni)
        acc[mi][ni] = __builtin_amdgcn_wmma_f32_16x16x32_bf16(
            false, a[mi].v, false, b[ni].v, (short)0, acc[mi][ni], false, false);
    __syncthreads();
  }

  // ---- epilogue ----
#pragma unroll
  for (int mi = 0; mi < MI; ++mi)
#pragma unroll
    for (int ni = 0; ni < NI; ++ni) {
      const int row0 = m0 + wm * WM + mi * 16 + laneHf * 8;
      const int col  = n0 + wn * WN + ni * 16 + laneLo;
      const float bv = bias ? bias[col] : 0.0f;
#pragma unroll
      for (int i = 0; i < 8; ++i) {
        const float vv = acc[mi][ni][i] * alpha + bv;
        if constexpr (OUT_BF16) {
          bf16_t* p = &((bf16_t*)Cb)[(long)(row0 + i) * ldc + col];
          if constexpr (NT_C) __builtin_nontemporal_store((bf16_t)vv, p);
          else *p = (bf16_t)vv;
        } else {
          float* p = &((float*)Cb)[(long)(row0 + i) * ldc + col];
          if constexpr (NT_C) __builtin_nontemporal_store(vv, p);
          else *p = vv;
        }
      }
    }
}

// ---------------------------------------------------------------------------
// Transpose-convert 4 fp32 DxD weights to bf16 W^T, packed consecutively.
// ---------------------------------------------------------------------------
__global__ __launch_bounds__(256)
void camsa_wconv(const float* __restrict__ wq, const float* __restrict__ wk,
                 const float* __restrict__ wv, const float* __restrict__ wo,
                 bf16_t* __restrict__ dst) {
  const float* srcs[4] = {wq, wk, wv, wo};
  const float* W = srcs[blockIdx.z];
  bf16_t* T = dst + (size_t)blockIdx.z * 1024 * 1024;
  __shared__ float tile[32][33];
  const int tx = threadIdx.x, ty = threadIdx.y;       // (32, 8)
  const int k0 = blockIdx.y * 32, n0 = blockIdx.x * 32;
#pragma unroll
  for (int i = 0; i < 4; ++i)
    tile[ty * 4 + i][tx] = W[(long)(k0 + ty * 4 + i) * 1024 + n0 + tx];
  __syncthreads();
#pragma unroll
  for (int i = 0; i < 4; ++i)
    T[(long)(n0 + ty * 4 + i) * 1024 + k0 + tx] = (bf16_t)tile[tx][ty * 4 + i];
}

// ---------------------------------------------------------------------------
// Context projection: ctx[b,:] = LN(c[b,:] @ Wc + bc) * g + be   (fp32, 8x1024)
// ---------------------------------------------------------------------------
__global__ __launch_bounds__(256)
void camsa_ctx_proj(const float* __restrict__ c, const float* __restrict__ Wc,
                    const float* __restrict__ bc, const float* __restrict__ g,
                    const float* __restrict__ be, float* __restrict__ outp) {
  const int b = blockIdx.x, tid = threadIdx.x;
  __shared__ float cs[256];
  __shared__ float s1[256], s2[256];
  cs[tid] = c[b * 256 + tid];
  __syncthreads();
  float acc[4], sum = 0.f, sq = 0.f;
#pragma unroll
  for (int j = 0; j < 4; ++j) {
    const int d = tid + j * 256;
    float a = bc[d];
    for (int f = 0; f < 256; ++f) a += cs[f] * Wc[(long)f * 1024 + d];
    acc[j] = a; sum += a; sq += a * a;
  }
  s1[tid] = sum; s2[tid] = sq; __syncthreads();
  for (int o = 128; o; o >>= 1) {
    if (tid < o) { s1[tid] += s1[tid + o]; s2[tid] += s2[tid + o]; }
    __syncthreads();
  }
  const float mu = s1[0] * (1.f / 1024.f);
  const float rs = rsqrtf(s2[0] * (1.f / 1024.f) - mu * mu + 1e-5f);
#pragma unroll
  for (int j = 0; j < 4; ++j) {
    const int d = tid + j * 256;
    outp[b * 1024 + d] = (acc[j] - mu) * rs * g[d] + be[d];
  }
}

// ---------------------------------------------------------------------------
// Row LN + bias + ctx add, pack bf16; dst and/or transposed vt may be null.
// One block per (b,s) row of 1024.
// ---------------------------------------------------------------------------
__global__ __launch_bounds__(256)
void camsa_ln_pack(const float* __restrict__ P, const float* __restrict__ bias,
                   const float* __restrict__ g, const float* __restrict__ be,
                   const float* __restrict__ ctxadd, bf16_t* __restrict__ dst,
                   bf16_t* __restrict__ vt) {
  const int rid = blockIdx.x;
  const int b = rid >> 10, s = rid & 1023;
  const int tid = threadIdx.x;
  const float* row = P + (long)rid * 1024;
  __shared__ float s1[256], s2[256];
  float x[4], sum = 0.f, sq = 0.f;
#pragma unroll
  for (int i = 0; i < 4; ++i) {
    const int d = tid + i * 256;
    x[i] = row[d] + bias[d];
    sum += x[i]; sq += x[i] * x[i];
  }
  s1[tid] = sum; s2[tid] = sq; __syncthreads();
  for (int o = 128; o; o >>= 1) {
    if (tid < o) { s1[tid] += s1[tid + o]; s2[tid] += s2[tid + o]; }
    __syncthreads();
  }
  const float mu = s1[0] * (1.f / 1024.f);
  const float rs = rsqrtf(s2[0] * (1.f / 1024.f) - mu * mu + 1e-5f);
#pragma unroll
  for (int i = 0; i < 4; ++i) {
    const int d = tid + i * 256;
    const float y = (x[i] - mu) * rs * g[d] + be[d] + ctxadd[b * 1024 + d];
    if (dst) dst[(long)rid * 1024 + d] = (bf16_t)y;
    if (vt) {
      const int h = d >> 6, dk = d & 63;
      vt[((long)((b * 16 + h) * 64 + dk)) * 1024 + s] = (bf16_t)y;
    }
  }
}

// ---------------------------------------------------------------------------
// In-place row softmax over last dim (1024), one block per row.  Fully
// streaming over 537MB -> non-temporal loads/stores.
// ---------------------------------------------------------------------------
__global__ __launch_bounds__(256)
void camsa_softmax(float* __restrict__ attn) {
  const long rid = blockIdx.x;
  float* row = attn + rid * 1024;
  const int tid = threadIdx.x;
  __shared__ float sm[256];
  float x[4], m = -3.4e38f;
#pragma unroll
  for (int i = 0; i < 4; ++i) {
    x[i] = __builtin_nontemporal_load(&row[tid + i * 256]);
    m = fmaxf(m, x[i]);
  }
  sm[tid] = m; __syncthreads();
  for (int o = 128; o; o >>= 1) {
    if (tid < o) sm[tid] = fmaxf(sm[tid], sm[tid + o]);
    __syncthreads();
  }
  m = sm[0]; __syncthreads();
  float sum = 0.f;
#pragma unroll
  for (int i = 0; i < 4; ++i) { x[i] = __expf(x[i] - m); sum += x[i]; }
  sm[tid] = sum; __syncthreads();
  for (int o = 128; o; o >>= 1) {
    if (tid < o) sm[tid] += sm[tid + o];
    __syncthreads();
  }
  const float inv = 1.f / sm[0];
#pragma unroll
  for (int i = 0; i < 4; ++i)
    __builtin_nontemporal_store(x[i] * inv, &row[tid + i * 256]);
}

// ---------------------------------------------------------------------------
extern "C" void kernel_launch(void* const* d_in, const int* in_sizes, int n_in,
                              void* d_out, int out_size, void* d_ws, size_t ws_size,
                              hipStream_t stream) {
  constexpr int B = 8, S = 1024, D = 1024, H = 16, DK = 64;
  constexpr float SCALE = 512.0f;   // float(D // H**-0.5)/sqrt(DK) = 4096/8

  const float* Q   = (const float*)d_in[0];
  const float* c   = (const float*)d_in[1];
  const float* Wq  = (const float*)d_in[2];  const float* bq  = (const float*)d_in[3];
  const float* Wk  = (const float*)d_in[4];  const float* bk  = (const float*)d_in[5];
  const float* Wv  = (const float*)d_in[6];  const float* bv  = (const float*)d_in[7];
  const float* Wcq = (const float*)d_in[8];  const float* bcq = (const float*)d_in[9];
  const float* Wck = (const float*)d_in[10]; const float* bck = (const float*)d_in[11];
  const float* Wcv = (const float*)d_in[12]; const float* bcv = (const float*)d_in[13];
  const float* g_q  = (const float*)d_in[14]; const float* be_q  = (const float*)d_in[15];
  const float* g_qc = (const float*)d_in[16]; const float* be_qc = (const float*)d_in[17];
  const float* g_k  = (const float*)d_in[18]; const float* be_k  = (const float*)d_in[19];
  const float* g_kc = (const float*)d_in[20]; const float* be_kc = (const float*)d_in[21];
  const float* g_v  = (const float*)d_in[22]; const float* be_v  = (const float*)d_in[23];
  const float* g_vc = (const float*)d_in[24]; const float* be_vc = (const float*)d_in[25];
  const float* Wo  = (const float*)d_in[26]; const float* bo  = (const float*)d_in[27];

  float* outp  = (float*)d_out;                       // (B,S,D)
  float* attnp = outp + (size_t)B * S * D;            // (B,H,S,S)

  // workspace carve-up (bytes, all 256-aligned)
  char* ws = (char*)d_ws;
  size_t off = 0;
  bf16_t* wt   = (bf16_t*)(ws + off); off += (size_t)4 * D * D * sizeof(bf16_t);   // Wq^T..Wo^T
  float*  ctxq = (float*)(ws + off);  off += (size_t)B * D * sizeof(float);
  float*  ctxk = (float*)(ws + off);  off += (size_t)B * D * sizeof(float);
  float*  ctxv = (float*)(ws + off);  off += (size_t)B * D * sizeof(float);
  float*  Pbuf = (float*)(ws + off);  off += (size_t)B * S * D * sizeof(float);    // pre-LN proj
  bf16_t* qb   = (bf16_t*)(ws + off); off += (size_t)B * S * D * sizeof(bf16_t);
  bf16_t* kb   = (bf16_t*)(ws + off); off += (size_t)B * S * D * sizeof(bf16_t);
  bf16_t* vt   = (bf16_t*)(ws + off); off += (size_t)B * S * D * sizeof(bf16_t);   // [b][h][dk][s]
  bf16_t* ctxb = (bf16_t*)(ws + off); off += (size_t)B * S * D * sizeof(bf16_t);
  (void)ws_size; (void)n_in; (void)in_sizes; (void)out_size;

  // 1) weights -> bf16 W^T
  camsa_wconv<<<dim3(32, 32, 4), dim3(32, 8), 0, stream>>>(Wq, Wk, Wv, Wo, wt);

  // 2) context projections (LN'd, fp32)
  camsa_ctx_proj<<<B, 256, 0, stream>>>(c, Wcq, bcq, g_qc, be_qc, ctxq);
  camsa_ctx_proj<<<B, 256, 0, stream>>>(c, Wck, bck, g_kc, be_kc, ctxk);
  camsa_ctx_proj<<<B, 256, 0, stream>>>(c, Wcv, bcv, g_vc, be_vc, ctxv);

  // 3) projections: P = Q @ W  (A fp32 on-the-fly bf16), then LN+pack
  const dim3 gProj(D / 128, (B * S) / 128, 1);
  camsa_gemm_nt<128, 128, true, false, false, false><<<gProj, 256, 0, stream>>>(
      Q, wt + 0 * (size_t)D * D, Pbuf, nullptr, D, D, D, D, 1, 0, 0, 0, 0, 0, 0, 1.0f);
  camsa_ln_pack<<<B * S, 256, 0, stream>>>(Pbuf, bq, g_q, be_q, ctxq, qb, nullptr);

  camsa_gemm_nt<128, 128, true, false, false, false><<<gProj, 256, 0, stream>>>(
      Q, wt + 1 * (size_t)D * D, Pbuf, nullptr, D, D, D, D, 1, 0, 0, 0, 0, 0, 0, 1.0f);
  camsa_ln_pack<<<B * S, 256, 0, stream>>>(Pbuf, bk, g_k, be_k, ctxk, kb, nullptr);

  camsa_gemm_nt<128, 128, true, false, false, false><<<gProj, 256, 0, stream>>>(
      Q, wt + 2 * (size_t)D * D, Pbuf, nullptr, D, D, D, D, 1, 0, 0, 0, 0, 0, 0, 1.0f);
  camsa_ln_pack<<<B * S, 256, 0, stream>>>(Pbuf, bv, g_v, be_v, ctxv, nullptr, vt);

  // 4) scores = SCALE * q @ k^T  per (b,h)  -> attn region (fp32, NT stores)
  camsa_gemm_nt<128, 128, false, false, false, true><<<dim3(S / 128, S / 128, B * H), 256, 0, stream>>>(
      qb, kb, attnp, nullptr, D, D, S, DK, H,
      (long)S * D, DK, (long)S * D, DK, (long)H * S * S, (long)S * S, SCALE);

  // 5) softmax rows (in-place, NT)
  camsa_softmax<<<B * H * S, 256, 0, stream>>>(attnp);

  // 6) ctx = attn @ v  per (b,h): A fp32 attn (in-register bf16, NT loads), B = v^T
  camsa_gemm_nt<128, 64, true, true, true, false><<<dim3(1, S / 128, B * H), 256, 0, stream>>>(
      attnp, vt, ctxb, nullptr, S, S, D, S, H,
      (long)H * S * S, (long)S * S, (long)H * DK * S, (long)DK * S, (long)S * D, DK, 1.0f);

  // 7) out = ctx @ Wo + bo
  camsa_gemm_nt<128, 128, false, false, false, false><<<gProj, 256, 0, stream>>>(
      ctxb, wt + 3 * (size_t)D * D, outp, bo, D, D, D, D, 1, 0, 0, 0, 0, 0, 0, 1.0f);
}
